// SimpleBlock_21723944583653
// MI455X (gfx1250) — matrix operand order repaired and checked by reading, not compiled, not run on hardware
//
#include <hip/hip_runtime.h>
#include <hip/hip_bf16.h>

// ---------------- types ----------------
typedef __attribute__((ext_vector_type(8)))  __bf16 v8bf;
typedef __attribute__((ext_vector_type(16))) __bf16 v16bf;
typedef __attribute__((ext_vector_type(8)))  float  v8f;

static __device__ __forceinline__ v16bf cat8(v8bf a, v8bf b) {
    return __builtin_shufflevector(a, b, 0,1,2,3,4,5,6,7,8,9,10,11,12,13,14,15);
}

#define NPTS_TOTAL_H 32
#define NKP 15
#define CIN 64
#define COUT 128
#define KDIM2 1024   // padded (k,c) contraction: 16*64, block k=15 is zero

// s_wait_asynccnt <= n  (ASYNCcnt tracks global_load_async_to_lds_*)
#define ASYNC_WAIT(n) asm volatile("s_wait_asynccnt " #n ::: "memory")

// ---------------- prep kernels ----------------
// weights fp32 [15][64][128] -> bf16 wT[d][kidx], kidx = k*64+c, padded to 1024.
// Also zeroes the 256-float stats area (block 0).
__global__ void prep_weights_kernel(const float* __restrict__ w,
                                    __bf16* __restrict__ wT,
                                    float* __restrict__ stats) {
    int i = blockIdx.x * blockDim.x + threadIdx.x;
    if (i < 256) stats[i] = 0.0f;
    if (i >= COUT * KDIM2) return;
    int d = i >> 10;
    int kidx = i & (KDIM2 - 1);
    float v = 0.0f;
    if (kidx < NKP * CIN) {
        v = w[(size_t)kidx * COUT + d];   // weights[(k*64+c)*128 + d]
    }
    wT[(size_t)d * KDIM2 + kidx] = (__bf16)v;
}

// x fp32 [N][64] -> bf16 (halves gather traffic, kills per-use converts)
__global__ void prep_x_kernel(const float* __restrict__ x,
                              __bf16* __restrict__ xbf, int n) {
    int i = blockIdx.x * blockDim.x + threadIdx.x;
    if (i < n) xbf[i] = (__bf16)x[i];
}

// ---------------- main fused KPConv kernel ----------------
// 4 waves / block, 16 points / wave, all matrix math via v_wmma_f32_16x16x32_bf16.
// Dynamic LDS (bytes), phase-overlapped:
//   [0,4096)          phase1: inflT 4w * [16][32] bf16   | phase2: wbuf0 (8KB) [0,8192)
//   [4096,20480)      phase1: nxT   4w * [64][32] bf16   | phase2: wbuf1 (8KB) [8192,16384)
//   [20480,151552)    wf : 4 waves * [16][1024] bf16 (A of phase 2)
//   [151552,152576)   sstats: 256 f32 block-partial BN sums
// Total 152576 B -> 2 blocks (8 waves) per 320KB WGP.
__global__ __launch_bounds__(128)
void kpconv_wmma_kernel(const float* __restrict__ points,
                        const int*   __restrict__ neighbors,
                        const __bf16* __restrict__ xbf,
                        const __bf16* __restrict__ wT,
                        const float* __restrict__ kpts,
                        float* __restrict__ out,
                        float* __restrict__ stats,
                        int n_points) {
    extern __shared__ char smem[];
    const int tid  = threadIdx.x;
    const int wave = tid >> 5;
    const int lane = tid & 31;
    const int m    = lane & 15;   // intra-halfwave id
    const int g    = lane >> 4;   // halfwave id

    __bf16* inflT  = (__bf16*)(smem)          + wave * (16 * 32);
    __bf16* nxT    = (__bf16*)(smem + 4096)   + wave * (CIN * 32);
    __bf16* wf     = (__bf16*)(smem + 20480)  + wave * (16 * KDIM2);
    float*  sstats = (float*)(smem + 151552);

    if (tid < 128) { sstats[tid] = 0.0f; sstats[tid + 128] = 0.0f; }
    __syncthreads();

    const int base = (blockIdx.x * 4 + wave) * 16;

    // kernel points: wave-uniform -> scalar loads
    float kx[NKP], ky[NKP], kz[NKP];
#pragma unroll
    for (int k = 0; k < NKP; ++k) {
        kx[k] = kpts[k * 3 + 0];
        ky[k] = kpts[k * 3 + 1];
        kz[k] = kpts[k * 3 + 2];
    }

    // pad row of A (k = 15) stays zero for all of phase 1
    inflT[15 * 32 + lane] = (__bf16)0.0f;

    // ---------------- phase 1: per point, wf[k][c] = infl^T @ nx ----------------
    for (int p = 0; p < 16; ++p) {
        int n = base + p;
        n = n < n_points ? n : (n_points - 1);

        // lane <-> neighbor h
        const int idx = neighbors[(size_t)n * NPTS_TOTAL_H + lane];
        const float cx = points[(size_t)n * 3 + 0];
        const float cy = points[(size_t)n * 3 + 1];
        const float cz = points[(size_t)n * 3 + 2];
        const float px = points[(size_t)idx * 3 + 0] - cx;
        const float py = points[(size_t)idx * 3 + 1] - cy;
        const float pz = points[(size_t)idx * 3 + 2] - cz;

#pragma unroll
        for (int k = 0; k < NKP; ++k) {
            const float dx = px - kx[k], dy = py - ky[k], dz = pz - kz[k];
            const float d2 = dx * dx + dy * dy + dz * dz;
            const float inf = fmaxf(0.0f, 1.0f - sqrtf(d2));  // KP_EXTENT = 1
            inflT[k * 32 + lane] = (__bf16)inf;               // A^T: [k][h]
        }

        // gather neighbor features, write transposed [c][h] for B fragments
        const v8bf* xr = (const v8bf*)(xbf + (size_t)idx * CIN);
#pragma unroll
        for (int j = 0; j < 8; ++j) {
            v8bf r = xr[j];
#pragma unroll
            for (int e = 0; e < 8; ++e) {
                nxT[(j * 8 + e) * 32 + lane] = r[e];
            }
        }

        // A fragment: row m, K contiguous 16 at g*16 (lanes 0-15: K 0-15; 16-31: K 16-31)
        v16bf A = *(const v16bf*)(inflT + m * 32 + g * 16);

#pragma unroll
        for (int t = 0; t < 4; ++t) {
            // B fragment: col n = t*16+m, K contiguous 16 at g*16 (col-major LDS)
            v16bf B = *(const v16bf*)(nxT + (t * 16 + m) * 32 + g * 16);
            v8f c = {0.f, 0.f, 0.f, 0.f, 0.f, 0.f, 0.f, 0.f};
            c = __builtin_amdgcn_wmma_f32_16x16x32_bf16(
                    false, A, false, B, (short)0, c, false, false);
            // D layout: lanes 0-15 VGPR r -> (M=r, N=lane); 16-31 -> (M=r+8, N=lane-16)
#pragma unroll
            for (int r = 0; r < 8; ++r) {
                const int kk = r + g * 8;
                const int cc = t * 16 + m;
                wf[p * KDIM2 + kk * CIN + cc] = (__bf16)c[r];  // row 15 == 0 by construction
            }
        }
    }

    // phase boundary: inflT/nxT are dead from here; their LDS becomes the
    // double-buffered weight tile (wbuf0 @ 0, wbuf1 @ 8192).
    __syncthreads();

    // ---- async stage of one 32-wide k-slice of wT into LDS ----
    // Tile = wT[d][ks*32 .. ks*32+31] for all 128 d -> 8KB, layout [d][32] bf16.
    // Each wave copies rows 32*wave..32*wave+31; 4 x b128 per lane, ASYNCcnt-tracked.
    const unsigned long long wT_base = (unsigned long long)(const void*)wT;
    const unsigned lds_base = (unsigned)(size_t)(void*)smem;
    const int row0  = wave * 32 + (lane >> 2);
    const int chunk = (lane & 3) * 16;   // byte chunk within a 64B row
#define STAGE_TILE(KS, BUF)                                                     \
    {                                                                           \
        const unsigned kbyte = (unsigned)(KS) * 64u;                            \
        _Pragma("unroll")                                                       \
        for (int i = 0; i < 4; ++i) {                                           \
            const int row = row0 + i * 8;                                       \
            const unsigned goff = (unsigned)row * 2048u + kbyte + (unsigned)chunk; \
            const unsigned ldso = lds_base + (unsigned)(BUF)*8192u +            \
                                  (unsigned)row * 64u + (unsigned)chunk;        \
            asm volatile("global_load_async_to_lds_b128 %0, %1, %2"             \
                         :: "v"(ldso), "v"(goff), "s"(wT_base) : "memory");     \
        }                                                                       \
    }

    STAGE_TILE(0, 0);

    // ---------------- phase 2: out[16][128] = wf[16][1024] @ wT^T ----------------
    v8f acc[8];
#pragma unroll
    for (int t = 0; t < 8; ++t) acc[t] = (v8f){0.f,0.f,0.f,0.f,0.f,0.f,0.f,0.f};

    for (int ks = 0; ks < 32; ++ks) {
        const int kb = ks * 32;
        if (ks + 1 < 32) {
            STAGE_TILE(ks + 1, (ks + 1) & 1);   // overlap copy of next slice
            ASYNC_WAIT(4);                      // in-order: current slice landed
        } else {
            ASYNC_WAIT(0);
        }
        __syncthreads();                        // all 4 waves' copies visible

        v16bf A = *(const v16bf*)(wf + m * KDIM2 + kb + g * 16);

        const __bf16* wb = (const __bf16*)(smem + (ks & 1) * 8192);
#pragma unroll
        for (int t = 0; t < 8; ++t) {
            v16bf B = *(const v16bf*)(wb + (t * 16 + m) * 32 + g * 16);
            acc[t] = __builtin_amdgcn_wmma_f32_16x16x32_bf16(
                         false, A, false, B, (short)0, acc[t], false, false);
        }
        __syncthreads();                        // reads done before buffer is re-staged
    }

    // ---------------- epilogue: store + BN partial sums ----------------
    const int valid = n_points - base;  // may be <=0 or >=16
#pragma unroll
    for (int t = 0; t < 8; ++t) {
        const int d = t * 16 + m;
        float s = 0.0f, s2 = 0.0f;
#pragma unroll
        for (int r = 0; r < 8; ++r) {
            const int p = r + g * 8;
            const float v = acc[t][r];
            if (p < valid) {
                out[(size_t)(base + p) * COUT + d] = v;
                s += v;
                s2 += v * v;
            }
        }
        atomicAdd(&sstats[d], s);
        atomicAdd(&sstats[COUT + d], s2);
    }
    __syncthreads();
    if (tid < 128) {
        atomicAdd(&stats[tid], sstats[tid]);
        atomicAdd(&stats[tid + COUT], sstats[tid + COUT]);
    }
}

// ---------------- BN finalize + LeakyReLU ----------------
__global__ void bn_lrelu_kernel(float* __restrict__ out,
                                const float* __restrict__ stats,
                                const float* __restrict__ gamma,
                                const float* __restrict__ beta,
                                int n_points) {
    const int i = blockIdx.x * blockDim.x + threadIdx.x;
    if (i >= n_points * COUT) return;
    const int d = i & (COUT - 1);
    const float invN = 1.0f / (float)n_points;
    const float mean = stats[d] * invN;
    const float var  = stats[COUT + d] * invN - mean * mean;
    float y = (out[i] - mean) * rsqrtf(var + 1e-5f) * gamma[d] + beta[d];
    out[i] = y > 0.0f ? y : 0.2f * y;
}

// ---------------- launch ----------------
extern "C" void kernel_launch(void* const* d_in, const int* in_sizes, int n_in,
                              void* d_out, int out_size, void* d_ws, size_t ws_size,
                              hipStream_t stream) {
    const float* points = (const float*)d_in[0];
    const float* x      = (const float*)d_in[1];
    const int*   neigh  = (const int*)  d_in[2];
    const float* kpts   = (const float*)d_in[3];
    const float* w      = (const float*)d_in[4];
    const float* gamma  = (const float*)d_in[5];
    const float* beta   = (const float*)d_in[6];
    float* out = (float*)d_out;

    const int n_points = in_sizes[0] / 3;
    const int n_xelems = in_sizes[1];          // N * 64

    // workspace layout
    char* ws = (char*)d_ws;
    float*  stats = (float*)ws;                               // 256 f32
    __bf16* wT    = (__bf16*)(ws + 1024);                     // 128*1024 bf16 = 256 KB
    __bf16* xbf   = (__bf16*)(ws + 1024 + COUT * KDIM2 * 2);  // N*64 bf16

    // 1) prep: weights transpose/convert + stats zero
    {
        int tot = COUT * KDIM2;
        prep_weights_kernel<<<(tot + 255) / 256, 256, 0, stream>>>(w, wT, stats);
    }
    // 2) prep: x -> bf16
    prep_x_kernel<<<(n_xelems + 255) / 256, 256, 0, stream>>>(x, xbf, n_xelems);

    // 3) fused KPConv (WMMA + async-to-LDS weight staging) + BN partials
    {
        const int pts_per_block = 64;  // 4 waves * 16 points
        const int blocks = (n_points + pts_per_block - 1) / pts_per_block;
        const size_t lds_bytes = 152576;
        kpconv_wmma_kernel<<<blocks, 128, lds_bytes, stream>>>(
            points, neigh, xbf, wT, kpts, out, stats, n_points);
    }

    // 4) BN finalize + LeakyReLU
    {
        int tot = n_points * COUT;
        bn_lrelu_kernel<<<(tot + 255) / 256, 256, 0, stream>>>(out, stats, gamma, beta, n_points);
    }
}
